// NBACliffordSharedSimplicialMPNN_50620484550970
// MI455X (gfx1250) — matrix operand order repaired and checked by reading, not compiled, not run on hardware
//
#include <hip/hip_runtime.h>
#include <cstdint>
#include <cstddef>

typedef __attribute__((ext_vector_type(16))) _Float16 v16h;
typedef __attribute__((ext_vector_type(8)))  float    v8f;
typedef __attribute__((ext_vector_type(4)))  int      i4v;

#define EPSV   1e-6f
#define SQRT2I 0.7071067811865476f

// ---- Cl(2,0) metadata (compile-time folded) ----
constexpr int   GBc[4]    = {0,1,1,2};
constexpr int   CAY_I[16] = {0,0,0,0, 1,1,1,1, 2,2,2,2, 3,3,3,3};
constexpr int   CAY_J[16] = {0,1,2,3, 1,0,3,2, 2,3,0,1, 3,2,1,0};
constexpr int   CAY_K[16] = {0,1,2,3, 0,1,2,3, 0,1,2,3, 0,1,2,3};
constexpr float CAY_S[16] = {1,1,1,1, 1,1,1,1, 1,-1,1,-1, 1,-1,1,-1};
constexpr int   P2v[2][2] = {{0,1},{1,0}};
constexpr int   P3v[6][3] = {{0,1,2},{0,2,1},{1,0,2},{1,2,0},{2,0,1},{2,1,0}};

__device__ __forceinline__ float sigm(float x){ return 1.f/(1.f+__expf(-x)); }

__device__ __forceinline__ v8f wmma16(v16h a, v16h b, v8f c){
  // v_wmma_f32_16x16x32_f16  (codegen-confirmed 8-arg signature)
  return __builtin_amdgcn_wmma_f32_16x16x32_f16(false, a, false, b, (short)0, c, false, false);
}

// A-fragment (16x32 f16): lanes 0-15 hold M=lane, K {0..7,16..23}; lanes 16-31 K {8..15,24..31}
__device__ __forceinline__ v16h afrag(const _Float16* base, int CP, int kt, int lane){
  int m  = lane & 15;
  int hi = (lane >> 4) & 1;
  const _Float16* p = base + m*CP + kt*32 + hi*8;
  v16h a;
#pragma unroll
  for(int t=0;t<8;t++){ a[t] = p[t]; a[8+t] = p[16+t]; }
  return a;
}

// B-fragment pre-packed as [g][kt][ot][lane][16] halves -> one aligned 32B load per lane
__device__ __forceinline__ v16h bfrag(const _Float16* pk, int g, int kt, int ot, int KT, int OT, int lane){
  return *reinterpret_cast<const v16h*>(pk + (size_t)(((g*KT + kt)*OT + ot)*32 + lane)*16);
}

// ---- CDNA5 async global->LDS staging (compile-guarded) ----
__device__ __forceinline__ void async_b128(const float* g, float* l){
#if defined(__gfx1250__) && __has_builtin(__builtin_amdgcn_global_load_async_to_lds_b128)
  __builtin_amdgcn_global_load_async_to_lds_b128(
      (__attribute__((address_space(1))) i4v*)(void*)g,
      (__attribute__((address_space(3))) i4v*)(void*)l, 0, 0);
#else
  *(float4*)l = *(const float4*)g;
#endif
}
__device__ __forceinline__ void async_wait_all(){
#if defined(__gfx1250__) && __has_builtin(__builtin_amdgcn_global_load_async_to_lds_b128)
#if __has_builtin(__builtin_amdgcn_s_wait_asynccnt)
  __builtin_amdgcn_s_wait_asynccnt(0);
#else
  asm volatile("s_wait_asynccnt 0" ::: "memory");
#endif
#endif
}

// ============ weight packer: f32 [O][C][3] (sub) or [O][C] (full) -> WMMA B layout f16 ============
__global__ void pack_weights_kernel(const float* __restrict__ W, _Float16* __restrict__ dst,
                                    int O, int C, int KT, int OT, int sub){
  int idx = blockIdx.x*blockDim.x + threadIdx.x;
  int total = 3*KT*OT*512;
  if(idx >= total) return;
  int t    = idx & 15;
  int lane = (idx >> 4) & 31;
  int rest = idx >> 9;
  int ot = rest % OT; rest /= OT;
  int kt = rest % KT;
  int g  = rest / KT;
  int K    = kt*32 + ((lane>>4)&1)*8 + (t & 7) + ((t >> 3) << 4);
  int Ncol = ot*16 + (lane & 15);
  float v = 0.f;
  if(K < C && Ncol < O)
    v = sub ? W[((size_t)Ncol*C + K)*3 + g] : W[(size_t)Ncol*C + K];
  dst[idx] = (_Float16)v;
}

// ============ full CEMLP block: mvlin_sub+bias -> mv_silu -> SGP -> mv_layernorm ============
template<int CIN, int COUT>
__global__ __launch_bounds__(32)
void cemlp_block_kernel(const float* __restrict__ in, float* __restrict__ out, int nRows,
                        const _Float16* __restrict__ pkW, const _Float16* __restrict__ pkWl,
                        const _Float16* __restrict__ pkWr,
                        const float* __restrict__ bb,  const float* __restrict__ bbl,
                        const float* __restrict__ sa,  const float* __restrict__ sb,
                        const float* __restrict__ na,  const float* __restrict__ gw,
                        const float* __restrict__ lnw){
  constexpr int KT1 = (CIN  + 31)/32, CINP  = KT1*32;
  constexpr int OT  = (COUT + 15)/16, COUTP = OT*16;
  constexpr int KT2 = (COUT + 31)/32, CUP   = KT2*32;

  __shared__ float    sRaw[16*CIN*4];
  __shared__ _Float16 sIn[4][16][CINP];
  __shared__ _Float16 sU [4][16][CUP];
  __shared__ float    sRes[16][COUTP][4];
  __shared__ float    sNorm[16];

  const int lane = threadIdx.x;
  const int n0   = blockIdx.x * 16;

  __builtin_prefetch(pkW, 0, 0);
  __builtin_prefetch(pkWl, 0, 0);
  __builtin_prefetch(pkWr, 0, 0);

  // ---- stage input tile (16 rows contiguous) into LDS ----
  const float* gbase = in + (size_t)n0*CIN*4;
  const int tileElems = 16*CIN*4;
  bool full = (n0 + 16) <= nRows;
  if(full){
    for(int ofs = lane*4; ofs < tileElems; ofs += 32*4)
      async_b128(gbase + ofs, &sRaw[ofs]);
    async_wait_all();
  } else {
    int valid = (nRows - n0)*CIN*4;
    for(int t = lane; t < tileElems; t += 32)
      sRaw[t] = (t < valid) ? gbase[t] : 0.f;
  }
  // convert to blade-major f16 A-tile
  for(int t = lane; t < 16*CINP; t += 32){
    int r = t / CINP, c = t % CINP;
#pragma unroll
    for(int i=0;i<4;i++)
      sIn[i][r][c] = (c < CIN) ? (_Float16)sRaw[(r*CIN + c)*4 + i] : (_Float16)0.f;
  }
  for(int t = lane; t < 4*16*CUP; t += 32) (&sU[0][0][0])[t] = (_Float16)0.f;
  __syncthreads();

  const int oo = (lane & 15);
  const int mb = 8*(lane >> 4);

  // ---- stage 1: t = W x + b ; u = mv_silu(t) -> sU ----
  for(int ot = 0; ot < OT; ++ot){
    v8f a0c = {}, a1c = {}, a2c = {}, a3c = {};
#pragma unroll
    for(int kt = 0; kt < KT1; ++kt){
      v16h A0 = afrag(&sIn[0][0][0], CINP, kt, lane);
      v16h A1 = afrag(&sIn[1][0][0], CINP, kt, lane);
      v16h A2 = afrag(&sIn[2][0][0], CINP, kt, lane);
      v16h A3 = afrag(&sIn[3][0][0], CINP, kt, lane);
      v16h B0 = bfrag(pkW, 0, kt, ot, KT1, OT, lane);
      v16h B1 = bfrag(pkW, 1, kt, ot, KT1, OT, lane);
      v16h B2 = bfrag(pkW, 2, kt, ot, KT1, OT, lane);
      a0c = wmma16(A0, B0, a0c);
      a1c = wmma16(A1, B1, a1c);
      a2c = wmma16(A2, B1, a2c);
      a3c = wmma16(A3, B2, a3c);
    }
    int o = ot*16 + oo;
    if(o < COUT){
      float bv  = bb[o];
      float sa0 = sa[o*3+0], sa1 = sa[o*3+1], sa2 = sa[o*3+2];
      float sb0 = sb[o*3+0], sb1 = sb[o*3+1], sb2 = sb[o*3+2];
#pragma unroll
      for(int r=0;r<8;r++){
        int m = mb + r;
        float x0 = a0c[r] + bv, x1 = a1c[r], x2 = a2c[r], x3 = a3c[r];
        float f0 = sigm(sa0*x0 + sb0);
        float f1 = sigm(sa1*(x1*x1 + x2*x2) + sb1);
        float f2 = sigm(sa2*(x3*x3) + sb2);
        sU[0][m][o] = (_Float16)(f0*x0);
        sU[1][m][o] = (_Float16)(f1*x1);
        sU[2][m][o] = (_Float16)(f1*x2);
        sU[3][m][o] = (_Float16)(f2*x3);
      }
    }
  }
  __syncthreads();

  // ---- stage 2: xl = Wl u + bl ; xr = norm_layer(Wr u) ; res = (xl + gp(u,xr)) * 1/sqrt2 ----
  for(int ot = 0; ot < OT; ++ot){
    v8f l0c={},l1c={},l2c={},l3c={}, r0c={},r1c={},r2c={},r3c={};
#pragma unroll
    for(int kt = 0; kt < KT2; ++kt){
      v16h A0 = afrag(&sU[0][0][0], CUP, kt, lane);
      v16h A1 = afrag(&sU[1][0][0], CUP, kt, lane);
      v16h A2 = afrag(&sU[2][0][0], CUP, kt, lane);
      v16h A3 = afrag(&sU[3][0][0], CUP, kt, lane);
      v16h L0 = bfrag(pkWl, 0, kt, ot, KT2, OT, lane);
      v16h L1 = bfrag(pkWl, 1, kt, ot, KT2, OT, lane);
      v16h L2 = bfrag(pkWl, 2, kt, ot, KT2, OT, lane);
      v16h R0 = bfrag(pkWr, 0, kt, ot, KT2, OT, lane);
      v16h R1 = bfrag(pkWr, 1, kt, ot, KT2, OT, lane);
      v16h R2 = bfrag(pkWr, 2, kt, ot, KT2, OT, lane);
      l0c = wmma16(A0, L0, l0c);
      l1c = wmma16(A1, L1, l1c);
      l2c = wmma16(A2, L1, l2c);
      l3c = wmma16(A3, L2, l3c);
      r0c = wmma16(A0, R0, r0c);
      r1c = wmma16(A1, R1, r1c);
      r2c = wmma16(A2, R1, r2c);
      r3c = wmma16(A3, R2, r3c);
    }
    int o = ot*16 + oo;
    if(o < COUT){
      float blv = bbl[o];
      float g0 = sigm(na[o*3+0]), g1 = sigm(na[o*3+1]), g2 = sigm(na[o*3+2]);
      const float* gwo = gw + (size_t)o*27;
#pragma unroll
      for(int r=0;r<8;r++){
        int m = mb + r;
        float q0 = r0c[r], q1 = r1c[r], q2 = r2c[r], q3 = r3c[r];
        float p0 = g0*(sqrtf(q0*q0) - 1.f) + 1.f;
        float p1 = g1*(sqrtf(q1*q1 + q2*q2) - 1.f) + 1.f;
        float p2 = g2*(sqrtf(q3*q3) - 1.f) + 1.f;
        float xr[4] = { q0/(p0+EPSV), q1/(p1+EPSV), q2/(p1+EPSV), q3/(p2+EPSV) };
        float uu[4] = { (float)sU[0][m][o], (float)sU[1][m][o],
                        (float)sU[2][m][o], (float)sU[3][m][o] };
        float gp[4] = {0.f,0.f,0.f,0.f};
#pragma unroll
        for(int t=0;t<16;t++)
          gp[CAY_J[t]] += CAY_S[t]*gwo[GBc[CAY_I[t]]*9 + GBc[CAY_J[t]]*3 + GBc[CAY_K[t]]]
                          * uu[CAY_I[t]] * xr[CAY_K[t]];
        sRes[m][o][0] = (l0c[r] + blv + gp[0]) * SQRT2I;
        sRes[m][o][1] = (l1c[r] + gp[1]) * SQRT2I;
        sRes[m][o][2] = (l2c[r] + gp[2]) * SQRT2I;
        sRes[m][o][3] = (l3c[r] + gp[3]) * SQRT2I;
      }
    }
  }
  __syncthreads();

  // ---- mv_layernorm: per-row mean over channels of |mv| ----
  if(lane < 16){
    float s = 0.f;
    for(int c = 0; c < COUT; ++c){
      float a0=sRes[lane][c][0], a1=sRes[lane][c][1], a2=sRes[lane][c][2], a3=sRes[lane][c][3];
      s += sqrtf(a0*a0 + a1*a1 + a2*a2 + a3*a3);
    }
    sNorm[lane] = s/(float)COUT + EPSV;
  }
  __syncthreads();
  for(int t = lane; t < 16*COUT*4; t += 32){
    int m = t/(COUT*4); int rem = t%(COUT*4); int c = rem/4;
    int row = n0 + m;
    if(row < nRows) out[(size_t)row*COUT*4 + rem] = lnw[c]*sRes[m][c][rem%4]/sNorm[m];
  }
}

// ============ plain multivector linear (bias on scalar blade) ============
template<int CIN, int COUT>
__global__ __launch_bounds__(32)
void mvlin_kernel(const float* __restrict__ in, float* __restrict__ out, int nRows,
                  const _Float16* __restrict__ pk, const float* __restrict__ bias){
  constexpr int KT = (CIN + 31)/32, CINP = KT*32;
  constexpr int OT = (COUT + 15)/16;
  __shared__ _Float16 sIn[4][16][CINP];
  const int lane = threadIdx.x;
  const int n0   = blockIdx.x * 16;
  __builtin_prefetch(pk, 0, 0);
  for(int t = lane; t < 16*CINP; t += 32){
    int r = t / CINP, c = t % CINP; int row = n0 + r;
#pragma unroll
    for(int i=0;i<4;i++)
      sIn[i][r][c] = (c < CIN && row < nRows) ? (_Float16)in[((size_t)row*CIN + c)*4 + i]
                                              : (_Float16)0.f;
  }
  __syncthreads();
  const int oo = (lane & 15);
  const int mb = 8*(lane >> 4);
  for(int ot = 0; ot < OT; ++ot){
    v8f a0c = {}, a1c = {}, a2c = {}, a3c = {};
#pragma unroll
    for(int kt = 0; kt < KT; ++kt){
      v16h A0 = afrag(&sIn[0][0][0], CINP, kt, lane);
      v16h A1 = afrag(&sIn[1][0][0], CINP, kt, lane);
      v16h A2 = afrag(&sIn[2][0][0], CINP, kt, lane);
      v16h A3 = afrag(&sIn[3][0][0], CINP, kt, lane);
      v16h B0 = bfrag(pk, 0, kt, ot, KT, OT, lane);
      v16h B1 = bfrag(pk, 1, kt, ot, KT, OT, lane);
      v16h B2 = bfrag(pk, 2, kt, ot, KT, OT, lane);
      a0c = wmma16(A0, B0, a0c);
      a1c = wmma16(A1, B1, a1c);
      a2c = wmma16(A2, B1, a2c);
      a3c = wmma16(A3, B2, a3c);
    }
    int o = ot*16 + oo;
    if(o < COUT){
      float bv = bias[o];
#pragma unroll
      for(int r=0;r<8;r++){
        int row = n0 + mb + r;
        if(row < nRows){
          float* po = out + ((size_t)row*COUT + o)*4;
          po[0] = a0c[r] + bv; po[1] = a1c[r]; po[2] = a2c[r]; po[3] = a3c[r];
        }
      }
    }
  }
}

// ============ glue kernels ============
__global__ void node_attr_kernel(int total, const int* __restrict__ nt,
                                 const float* __restrict__ temb, float* __restrict__ na){
  int idx = blockIdx.x*blockDim.x + threadIdx.x;
  if(idx >= total) return;                 // total = N*3
  int n = idx/3, k = idx%3;
  float* o = na + (size_t)idx*4;
  o[0] = temb[nt[n]*3 + k]; o[1]=0.f; o[2]=0.f; o[3]=0.f;
}

__global__ void feat0_kernel(int total, const float* __restrict__ pos, const float* __restrict__ vel,
                             const int* __restrict__ x_ind, float* __restrict__ out, int F){
  int idx = blockIdx.x*blockDim.x + threadIdx.x;
  if(idx >= total) return;                 // total = B*6 * 2F
  int CH = 2*F;
  int c = idx % CH, row = idx / CH;
  int b = row/6, l = row%6;
  int pt = x_ind[(b*41 + l)*3 + 0] + b*6;
  int f = c % F;
  const float* src = (c >= F) ? vel : pos;
  float c0 = src[((size_t)pt*F + f)*2 + 0], c1 = src[((size_t)pt*F + f)*2 + 1];
  float* o = out + ((size_t)row*CH + c)*4;
  o[0]=0.f; o[1]=c0; o[2]=c1; o[3]=0.f;
}

__global__ void feat1_kernel(int total, const float* __restrict__ pos, const float* __restrict__ vel,
                             const int* __restrict__ x_ind, float* __restrict__ out, int F){
  int idx = blockIdx.x*blockDim.x + threadIdx.x;
  if(idx >= total) return;                 // total = B*15*2 * 4F
  int CH = 4*F;
  int c = idx % CH, row = idx / CH;
  int p = row % 2, n = row / 2;
  int b = n/15, l = n%15;
  bool isv = (c >= 2*F); int cc = isv ? c - 2*F : c;
  int v = cc / F, f = cc % F;
  int pt = x_ind[(b*41 + 6 + l)*3 + P2v[p][v]] + b*6;
  const float* src = isv ? vel : pos;
  float c0 = src[((size_t)pt*F + f)*2 + 0], c1 = src[((size_t)pt*F + f)*2 + 1];
  float* o = out + ((size_t)row*CH + c)*4;
  o[0]=0.f; o[1]=c0; o[2]=c1; o[3]=0.f;
}

__global__ void feat2_kernel(int total, const float* __restrict__ pos, const float* __restrict__ vel,
                             const int* __restrict__ x_ind, float* __restrict__ out, int F){
  int idx = blockIdx.x*blockDim.x + threadIdx.x;
  if(idx >= total) return;                 // total = B*20*6 * 6F
  int CH = 6*F;
  int c = idx % CH, row = idx / CH;
  int p = row % 6, n = row / 6;
  int b = n/20, l = n%20;
  bool isv = (c >= 3*F); int cc = isv ? c - 3*F : c;
  int v = cc / F, f = cc % F;
  int pt = x_ind[(b*41 + 21 + l)*3 + P3v[p][v]] + b*6;
  const float* src = isv ? vel : pos;
  float c0 = src[((size_t)pt*F + f)*2 + 0], c1 = src[((size_t)pt*F + f)*2 + 1];
  float* o = out + ((size_t)row*CH + c)*4;
  o[0]=0.f; o[1]=c0; o[2]=c1; o[3]=0.f;
}

__global__ void permsum_kernel(int total, const float* __restrict__ in, float* __restrict__ out,
                               int P, int CH){
  int idx = blockIdx.x*blockDim.x + threadIdx.x;
  if(idx >= total) return;                 // total = nOut*CH
  int n = idx / CH, t = idx % CH;
  float s = 0.f;
  for(int p=0;p<P;p++) s += in[((size_t)(n*P + p))*CH + t];
  out[idx] = s;
}

__global__ void fein_kernel(int total, const float* __restrict__ e0, const float* __restrict__ e1,
                            const float* __restrict__ e2, const float* __restrict__ na,
                            float* __restrict__ out){
  int idx = blockIdx.x*blockDim.x + threadIdx.x;
  if(idx >= total) return;                 // total = N*23*4
  int t = idx % 92, n = idx / 92;
  int b = n/41, l = n%41;
  int c = t/4, i = t%4;
  float v;
  if(c < 20){
    const float* src; int row;
    if(l < 6)      { src = e0; row = b*6  + l;      }
    else if(l < 21){ src = e1; row = b*15 + l - 6;  }
    else           { src = e2; row = b*20 + l - 21; }
    v = src[((size_t)row*20 + c)*4 + i];
  } else v = na[((size_t)n*3 + (c - 20))*4 + i];
  out[idx] = v;
}

__global__ void edge_gather_kernel(int total, const float* __restrict__ h,
                                   const float* __restrict__ na, const int* __restrict__ ei,
                                   float* __restrict__ out, int E){
  int idx = blockIdx.x*blockDim.x + threadIdx.x;
  if(idx >= total) return;                 // total = E*86*4
  int t = idx % 344, e = idx / 344;
  int c = t/4, i = t%4;
  int s = ei[e], d = ei[E + e];            // s = ei[0] (source), d = ei[1] (target)
  float v;
  if(c < 40)      v = h[((size_t)d*40 + c)*4 + i];
  else if(c < 80) v = h[((size_t)s*40 + (c-40))*4 + i];
  else if(c < 83) v = na[((size_t)s*3 + (c-80))*4 + i];
  else            v = na[((size_t)d*3 + (c-83))*4 + i];
  out[idx] = v;
}

__global__ void zero_kernel(int total, float* __restrict__ p){
  int idx = blockIdx.x*blockDim.x + threadIdx.x;
  if(idx < total) p[idx] = 0.f;
}

__global__ void segsum_kernel(int total, const float* __restrict__ m, const int* __restrict__ ei,
                              float* __restrict__ agg, int E){
  int idx = blockIdx.x*blockDim.x + threadIdx.x;
  if(idx >= total) return;                 // total = E*160
  int e = idx/160, t = idx%160;
  atomicAdd(&agg[(size_t)ei[E + e]*160 + t], m[idx]);
}

__global__ void node_gather_kernel(int total, const float* __restrict__ h,
                                   const float* __restrict__ agg, const float* __restrict__ na,
                                   float* __restrict__ out){
  int idx = blockIdx.x*blockDim.x + threadIdx.x;
  if(idx >= total) return;                 // total = N*83*4
  int t = idx % 332, n = idx / 332;
  int c = t/4, i = t%4;
  float v;
  if(c < 40)      v = h[((size_t)n*40 + c)*4 + i];
  else if(c < 80) v = agg[((size_t)n*40 + (c-40))*4 + i];
  else            v = na[((size_t)n*3 + (c-80))*4 + i];
  out[idx] = v;
}

__global__ void residual_kernel(int total, float* __restrict__ h, const float* __restrict__ u){
  int idx = blockIdx.x*blockDim.x + threadIdx.x;
  if(idx < total) h[idx] += u[idx];
}

__global__ void rd_gather_kernel(int total, const float* __restrict__ h, float* __restrict__ out){
  int idx = blockIdx.x*blockDim.x + threadIdx.x;
  if(idx >= total) return;                 // total = B*6*160
  int row = idx/160;
  int b = row/6, l = row%6;
  out[idx] = h[((size_t)(b*41 + l))*160 + idx%160];
}

__global__ void readout_kernel(const float* __restrict__ pout, const float* __restrict__ y,
                               float* __restrict__ dout, int B){
  int b = blockIdx.x, tid = threadIdx.x;   // 64 threads
  __shared__ float sa_[64], sf_[64];
  float asum = 0.f, fsum = 0.f;
  for(int t = tid; t < 5*40; t += 64){
    int pl = t/40, c = t%40;
    const float* po = pout + ((size_t)(b*6 + pl)*40 + c)*4;
    const float* yo = y    + ((size_t)(b*5 + pl)*40 + c)*2;
    float dx = po[1] - yo[0], dy = po[2] - yo[1];
    float d = sqrtf(dx*dx + dy*dy);
    asum += d;
    if(c == 39) fsum += d;
  }
  sa_[tid] = asum; sf_[tid] = fsum; __syncthreads();
  for(int s = 32; s > 0; s >>= 1){
    if(tid < s){ sa_[tid] += sa_[tid+s]; sf_[tid] += sf_[tid+s]; }
    __syncthreads();
  }
  if(tid == 0){ dout[1 + b] = sa_[0]/200.f; dout[1 + B + b] = sf_[0]/5.f; }
}

__global__ void final_mean_kernel(float* __restrict__ dout, int B){
  __shared__ float s_[256];
  float acc = 0.f;
  for(int i = threadIdx.x; i < B; i += 256) acc += dout[1 + i];
  s_[threadIdx.x] = acc; __syncthreads();
  for(int st = 128; st > 0; st >>= 1){
    if(threadIdx.x < st) s_[threadIdx.x] += s_[threadIdx.x + st];
    __syncthreads();
  }
  if(threadIdx.x == 0) dout[0] = s_[0]/(float)B;
}

// ============ host helpers ============
struct Ctx { char* base; size_t off; hipStream_t s; };

static void* ws_alloc(Ctx& c, size_t bytes){
  c.off = (c.off + 255) & ~(size_t)255;
  void* p = c.base + c.off;
  c.off += bytes;
  return p;
}
static float* allocf(Ctx& c, size_t nElem){ return (float*)ws_alloc(c, nElem*sizeof(float)); }

static const _Float16* pack(Ctx& c, const float* W, int O, int C, bool sub){
  int KT = (C + 31)/32, OT = (O + 15)/16;
  size_t n = (size_t)3*KT*OT*512;
  _Float16* d = (_Float16*)ws_alloc(c, n*sizeof(_Float16));
  pack_weights_kernel<<<(int)((n + 255)/256), 256, 0, c.s>>>(W, d, O, C, KT, OT, sub ? 1 : 0);
  return d;
}

struct RBlk { const float *W,*Wl,*Wr,*b,*bl,*gw,*ln,*na,*sa,*sb; };
struct PBlk { const _Float16 *W,*Wl,*Wr; const float *b,*bl,*sa,*sb,*na,*gw,*ln; };

template<int CIN, int COUT>
static void run_block(hipStream_t s, const float* in, float* out, int nRows, const PBlk& p){
  cemlp_block_kernel<CIN,COUT><<<(nRows + 15)/16, 32, 0, s>>>(
      in, out, nRows, p.W, p.Wl, p.Wr, p.b, p.bl, p.sa, p.sb, p.na, p.gw, p.ln);
}
template<int CIN, int COUT>
static void run_linear(hipStream_t s, const float* in, float* out, int nRows,
                       const _Float16* pk, const float* bias){
  mvlin_kernel<CIN,COUT><<<(nRows + 15)/16, 32, 0, s>>>(in, out, nRows, pk, bias);
}

static inline int gblk(long long total){ return (int)((total + 255)/256); }

extern "C" void kernel_launch(void* const* d_in, const int* in_sizes, int n_in,
                              void* d_out, int out_size, void* d_ws, size_t ws_size,
                              hipStream_t stream){
  (void)n_in; (void)out_size; (void)ws_size;
  const float* pos = (const float*)d_in[0];
  const float* vel = (const float*)d_in[1];
  const float* yy  = (const float*)d_in[2];
  const int* node_types = (const int*)d_in[3];
  const int* x_ind      = (const int*)d_in[4];
  const int* edge_index = (const int*)d_in[5];

  const int N = in_sizes[3];
  const int B = N / 41;
  const int F = in_sizes[0] / (6*B*2);
  const int E = in_sizes[5] / 2;

  // ---- params: jax-tree (alphabetical) leaf order ----
  int pi = 6;
  auto F32  = [&](){ return (const float*)d_in[pi++]; };
  auto rblk = [&](){
    RBlk r;
    r.W = F32(); r.Wl = F32(); r.Wr = F32(); r.b = F32(); r.bl = F32();
    r.gw = F32(); r.ln = F32(); r.na = F32(); r.sa = F32(); r.sb = F32();
    return r;
  };

  Ctx ctx{(char*)d_ws, 0, stream};
  auto pblk = [&](const RBlk& r, int CIN, int COUT){
    PBlk p;
    p.W  = pack(ctx, r.W,  COUT, CIN,  true);
    p.Wl = pack(ctx, r.Wl, COUT, COUT, true);
    p.Wr = pack(ctx, r.Wr, COUT, COUT, true);
    p.b = r.b; p.bl = r.bl; p.sa = r.sa; p.sb = r.sb; p.na = r.na; p.gw = r.gw; p.ln = r.ln;
    return p;
  };

  const float* cl0W = F32(); const float* cl0b = F32();
  PBlk cl1p  = pblk(rblk(), 40, 20);
  PBlk cl2ap = pblk(rblk(), 60, 40);
  PBlk cl2bp = pblk(rblk(), 40, 20);
  const float* feW = F32(); const float* feb = F32();
  PBlk Le1[4], Le2[4], Ln1[4], Ln2[4];
  for(int l=0;l<4;l++){
    Le1[l] = pblk(rblk(), 86, 40);
    Le2[l] = pblk(rblk(), 40, 40);
    Ln1[l] = pblk(rblk(), 83, 40);
    Ln2[l] = pblk(rblk(), 40, 40);
  }
  const float* pWw = F32(); const float* pbb = F32();
  const float* type_emb = F32();

  const _Float16* pk_cl0 = pack(ctx, cl0W, 20, 20, false);
  const _Float16* pk_fe  = pack(ctx, feW,  40, 23, false);
  const _Float16* pk_pW  = pack(ctx, pWw,  40, 40, true);

  // ---- activation buffers ----
  float* nattr = allocf(ctx, (size_t)N*12);
  float* e0in  = allocf(ctx, (size_t)B*6*80);
  float* e0out = allocf(ctx, (size_t)B*6*80);
  float* f1    = allocf(ctx, (size_t)B*30*160);
  float* f1out = allocf(ctx, (size_t)B*30*80);
  float* e1out = allocf(ctx, (size_t)B*15*80);
  float* f2    = allocf(ctx, (size_t)B*120*240);
  float* f2mid = allocf(ctx, (size_t)B*120*160);
  float* f2out = allocf(ctx, (size_t)B*120*80);
  float* e2out = allocf(ctx, (size_t)B*20*80);
  float* fein  = allocf(ctx, (size_t)N*92);
  float* h     = allocf(ctx, (size_t)N*160);
  float* minb  = allocf(ctx, (size_t)E*344);
  float* mmid  = allocf(ctx, (size_t)E*160);
  float* mout  = allocf(ctx, (size_t)E*160);
  float* agg   = allocf(ctx, (size_t)N*160);
  float* ninb  = allocf(ctx, (size_t)N*332);
  float* nmid  = allocf(ctx, (size_t)N*160);
  float* nupd  = allocf(ctx, (size_t)N*160);
  float* oin   = allocf(ctx, (size_t)B*6*160);
  float* pout  = allocf(ctx, (size_t)B*6*160);

  // ---- embeddings ----
  node_attr_kernel<<<gblk((long long)N*3),256,0,stream>>>(N*3, node_types, type_emb, nattr);

  feat0_kernel<<<gblk((long long)B*6*2*F),256,0,stream>>>(B*6*2*F, pos, vel, x_ind, e0in, F);
  run_linear<20,20>(stream, e0in, e0out, B*6, pk_cl0, cl0b);

  feat1_kernel<<<gblk((long long)B*30*4*F),256,0,stream>>>(B*30*4*F, pos, vel, x_ind, f1, F);
  run_block<40,20>(stream, f1, f1out, B*30, cl1p);
  permsum_kernel<<<gblk((long long)B*15*80),256,0,stream>>>(B*15*80, f1out, e1out, 2, 80);

  feat2_kernel<<<gblk((long long)B*120*6*F),256,0,stream>>>(B*120*6*F, pos, vel, x_ind, f2, F);
  run_block<60,40>(stream, f2, f2mid, B*120, cl2ap);
  run_block<40,20>(stream, f2mid, f2out, B*120, cl2bp);
  permsum_kernel<<<gblk((long long)B*20*80),256,0,stream>>>(B*20*80, f2out, e2out, 6, 80);

  fein_kernel<<<gblk((long long)N*92),256,0,stream>>>(N*92, e0out, e1out, e2out, nattr, fein);
  run_linear<23,40>(stream, fein, h, N, pk_fe, feb);

  // ---- 4 EGCL layers ----
  for(int l=0;l<4;l++){
    edge_gather_kernel<<<gblk((long long)E*344),256,0,stream>>>(E*344, h, nattr, edge_index, minb, E);
    run_block<86,40>(stream, minb, mmid, E, Le1[l]);
    run_block<40,40>(stream, mmid, mout, E, Le2[l]);
    zero_kernel<<<gblk((long long)N*160),256,0,stream>>>(N*160, agg);
    segsum_kernel<<<gblk((long long)E*160),256,0,stream>>>(E*160, mout, edge_index, agg, E);
    node_gather_kernel<<<gblk((long long)N*332),256,0,stream>>>(N*332, h, agg, nattr, ninb);
    run_block<83,40>(stream, ninb, nmid, N, Ln1[l]);
    run_block<40,40>(stream, nmid, nupd, N, Ln2[l]);
    residual_kernel<<<gblk((long long)N*160),256,0,stream>>>(N*160, h, nupd);
  }

  // ---- readout ----
  rd_gather_kernel<<<gblk((long long)B*6*160),256,0,stream>>>(B*6*160, h, oin);
  run_linear<40,40>(stream, oin, pout, B*6, pk_pW, pbb);
  readout_kernel<<<B, 64, 0, stream>>>(pout, yy, (float*)d_out, B);
  final_mean_kernel<<<1, 256, 0, stream>>>((float*)d_out, B);
}
